// RouterCompound_1984274891213
// MI455X (gfx1250) — matrix operand order repaired and checked by reading, not compiled, not run on hardware
//
#include <hip/hip_runtime.h>
#include <math.h>

// CDNA5 wave32 WMMA types
typedef __attribute__((ext_vector_type(16))) __bf16 v16bf;
typedef __attribute__((ext_vector_type(8)))  __bf16 v8bf;
typedef __attribute__((ext_vector_type(8)))  float  v8f;
typedef __attribute__((ext_vector_type(4)))  float  v4f;

#define DIMK   2048
#define NEXP   8
#define INNER  16
#define NT     9             // N tiles of 16: tile0 = W_out (8 cols + 8 pad), tiles 1..8 = experts
#define NCOL   (NT * 16)     // 144 combined gate columns
#define KC     128           // K chunk staged in LDS
#define KSTEP  32            // K per bf16 WMMA
#define PITCH  (KC + 8)      // bf16 elems per LDS row: 272B rows -> 16B aligned, conflict-free
#define PLANE  (NCOL * PITCH)// hi plane / lo plane stride (bf16 elems)
#define WAVES  8             // waves per block; each wave owns one 16-token M-tile
#define ROWP   (NCOL + 1)    // padded row for the f32 D dump (reuses same LDS)

__global__ __launch_bounds__(WAVES * 32)
void router_moe_kernel(const float* __restrict__ x,
                       const float* __restrict__ W_out,
                       const float* __restrict__ b_out,
                       const float* __restrict__ W_in,
                       const float* __restrict__ b_in,
                       float* __restrict__ out,
                       int bs)
{
    extern __shared__ float smem[];          // max(2*PLANE bf16, WAVES*16*ROWP f32)
    __bf16* sb = (__bf16*)smem;

    const int tid  = threadIdx.x;
    const int lane = tid & 31;
    const int wv   = tid >> 5;
    const int m    = lane & 15;              // A row / B column within tile
    const int kh   = lane >> 4;              // K half-select
    const int mtile   = blockIdx.x * WAVES + wv;
    const int tokBase = mtile * 16;

    v8f acc[NT] = {};                        // 9 x (16x16 f32) accumulators = 72 VGPRs

    // bf16 A 16x32 fragment: lane holds K runs [8kh, 8kh+8) and [16+8kh, 16+8kh+8)
    const float* aptr = x + (size_t)(tokBase + m) * DIMK + 8 * kh;

    for (int k0 = 0; k0 < DIMK; k0 += KC) {
        // ---- cooperative stage of W chunk into LDS as hi/lo bf16 planes ----
        for (int i = tid; i < NCOL * KC; i += WAVES * 32) {
            int c = i % NCOL;                // consecutive lanes -> consecutive cols (coalesced)
            int k = i / NCOL;
            float w;
            if (c < 16) {
                w = (c < NEXP) ? W_out[(size_t)(k0 + k) * NEXP + c] : 0.0f;
            } else {
                int e = (c - 16) >> 4;
                int j = (c - 16) & 15;
                w = W_in[((size_t)e * DIMK + (k0 + k)) * INNER + j];
            }
            __bf16 h = (__bf16)w;                         // RNE
            __bf16 l = (__bf16)(w - (float)h);            // residual
            sb[c * PITCH + k]         = h;
            sb[PLANE + c * PITCH + k] = l;
        }
        __syncthreads();

        // ---- K=32 steps: split-bf16 product xh*Wh + xl*Wh + xh*Wl (f32 acc) ----
        #pragma unroll
        for (int st = 0; st < KC / KSTEP; ++st) {
            const float* ap = aptr + k0 + st * KSTEP;
            float fa[16];
            *(v4f*)&fa[0]  = *(const v4f*)(ap);
            *(v4f*)&fa[4]  = *(const v4f*)(ap + 4);
            *(v4f*)&fa[8]  = *(const v4f*)(ap + 16);
            *(v4f*)&fa[12] = *(const v4f*)(ap + 20);
            v16bf ah, al;
            #pragma unroll
            for (int i = 0; i < 16; ++i) {
                __bf16 h = (__bf16)fa[i];
                ah[i] = h;
                al[i] = (__bf16)(fa[i] - (float)h);
            }
            // bf16 B 32x16 fragment: lane holds K runs [8kh, 8kh+8) and [16+8kh, 16+8kh+8)
            const int kb = st * KSTEP + 8 * kh;
            #pragma unroll
            for (int t = 0; t < NT; ++t) {
                const __bf16* p = sb + (t * 16 + m) * PITCH + kb;
                v8bf bh0 = *(const v8bf*)(p);
                v8bf bh1 = *(const v8bf*)(p + 16);
                v8bf bl0 = *(const v8bf*)(p + PLANE);
                v8bf bl1 = *(const v8bf*)(p + PLANE + 16);
                v16bf bh = __builtin_shufflevector(bh0, bh1,
                    0,1,2,3,4,5,6,7,8,9,10,11,12,13,14,15);
                v16bf bl = __builtin_shufflevector(bl0, bl1,
                    0,1,2,3,4,5,6,7,8,9,10,11,12,13,14,15);
                acc[t] = __builtin_amdgcn_wmma_f32_16x16x32_bf16(
                             false, ah, false, bh, (short)0, acc[t], false, false);
                acc[t] = __builtin_amdgcn_wmma_f32_16x16x32_bf16(
                             false, al, false, bh, (short)0, acc[t], false, false);
                acc[t] = __builtin_amdgcn_wmma_f32_16x16x32_bf16(
                             false, ah, false, bl, (short)0, acc[t], false, false);
            }
        }
        __syncthreads();
    }

    // ---- dump D tiles to (reused) LDS: rows = token-in-tile, cols = gate column ----
    float* dbase = smem + wv * (16 * ROWP);
    #pragma unroll
    for (int t = 0; t < NT; ++t)
        #pragma unroll
        for (int r = 0; r < 8; ++r)
            dbase[(r + 8 * kh) * ROWP + t * 16 + m] = acc[t][r];
    __syncthreads();

    // ---- per-token routing: softmax monotonic -> top-k on raw logits;
    //      normalized top-2 weights need only the two logits (Z cancels). ----
    if (lane < 16) {
        const float* row = dbase + lane * ROWP;
        const int tok = tokBase + lane;

        float ol[NEXP];
        #pragma unroll
        for (int c = 0; c < NEXP; ++c) ol[c] = row[c] + b_out[c];

        int e0 = 0;
        #pragma unroll
        for (int c = 1; c < NEXP; ++c) if (ol[c] > ol[e0]) e0 = c;
        int e1 = 0; float l1 = -INFINITY;
        #pragma unroll
        for (int c = 0; c < NEXP; ++c)
            if (c != e0 && ol[c] > l1) { l1 = ol[c]; e1 = c; }

        float s   = __expf(l1 - ol[e0]);     // <= 1, numerically stable
        float rw0 = 1.0f / (1.0f + s);
        float rw1 = s * rw0;

        float* wout = out + (size_t)tok * 6;
        float* iout = out + (size_t)bs * 6 + (size_t)tok * 6;
        wout[0] = rw0; wout[1] = rw0; wout[2] = rw0; wout[3] = rw0;
        wout[4] = rw1; wout[5] = rw1;

        int opos = 0;
        const int sel[2]  = { e0, e1 };
        const int kcnt[2] = { 4, 2 };
        #pragma unroll
        for (int j = 0; j < 2; ++j) {
            const int e = sel[j];
            float v[INNER];
            #pragma unroll
            for (int i = 0; i < INNER; ++i)
                v[i] = row[16 + e * INNER + i] + b_in[e * INNER + i];
            unsigned taken = 0u;
            for (int r = 0; r < kcnt[j]; ++r) {
                int best = 0; float bv = -INFINITY;
                #pragma unroll
                for (int i = 0; i < INNER; ++i)
                    if (!((taken >> i) & 1u) && v[i] > bv) { bv = v[i]; best = i; }
                taken |= 1u << best;
                iout[opos++] = (float)(e * INNER + best);
            }
        }
    }
}

extern "C" void kernel_launch(void* const* d_in, const int* in_sizes, int n_in,
                              void* d_out, int out_size, void* d_ws, size_t ws_size,
                              hipStream_t stream)
{
    (void)n_in; (void)out_size; (void)d_ws; (void)ws_size;
    const float* x     = (const float*)d_in[0];
    const float* W_out = (const float*)d_in[1];
    const float* b_out = (const float*)d_in[2];
    const float* W_in  = (const float*)d_in[3];
    const float* b_in  = (const float*)d_in[4];
    float* out = (float*)d_out;

    const int bs = in_sizes[0] / DIMK;                 // 16384
    const int grid = bs / (WAVES * 16);                // 128 blocks of 8 waves

    const size_t smemW = (size_t)2 * PLANE * sizeof(__bf16);            // 78,336 B
    const size_t smemD = (size_t)WAVES * 16 * ROWP * sizeof(float);     // 74,240 B
    const size_t smemBytes = smemW > smemD ? smemW : smemD;             // < 320 KB

    router_moe_kernel<<<grid, WAVES * 32, smemBytes, stream>>>(
        x, W_out, b_out, W_in, b_in, out, bs);
}
